// DeepVONet_2954937499697
// MI455X (gfx1250) — compile-verified, matrix operand order checked by reading, not run on hardware
//
#include <hip/hip_runtime.h>
#include <hip/hip_bf16.h>
#include <stddef.h>

typedef __attribute__((ext_vector_type(16))) __bf16 v16bf;
typedef __attribute__((ext_vector_type(8)))  __bf16 v8bf;
typedef __attribute__((ext_vector_type(8)))  float  v8f;

#define T_STEPS 128
#define BATCH   64
#define IN_DIM  1024
#define HID     1000
#define HPAD    1024
#define GATES   4000
#define NT      250              // 4000 / 16 N-tiles
#define KT      64               // 2048 / 32 K-tiles
#define FRAG_ELEMS 512           // 32x16 bf16 per fragment
#define PACK_ELEMS ((size_t)NT * KT * FRAG_ELEMS)   // 8,192,000

// ---------------------------------------------------------------------------
// Pack [W ; U] (f32, gate-major columns) into bf16 WMMA B-fragment order with
// gate-interleaved columns (col' = 4*j + gate) and zero-padded K rows.
// Fragment (nt, kt): 32 lanes x 16 bf16, lane n-column mapping per CDNA5 ISA:
//   lanes 0-15  -> n = lane,    k_local = {0..7, 16..23}
//   lanes 16-31 -> n = lane-16, k_local = {8..15, 24..31}
// ---------------------------------------------------------------------------
__global__ void pack_wu_kernel(const float* __restrict__ W, int wRows,
                               const float* __restrict__ U, int uRows,
                               __bf16* __restrict__ dst) {
    size_t e = (size_t)blockIdx.x * 256 + threadIdx.x;
    if (e >= PACK_ELEMS) return;
    int within = (int)(e & 511);
    int frag   = (int)(e >> 9);
    int kt     = frag & 63;
    int nt     = frag >> 6;
    int lane   = within >> 4;
    int v      = within & 15;
    int n      = nt * 16 + (lane & 15);
    int klocal = ((lane < 16) ? 0 : 8) + ((v >> 3) << 4) + (v & 7);
    int k      = kt * 32 + klocal;
    int j = n >> 2, gate = n & 3;
    int col = gate * HID + j;            // original gate-major column
    float val = 0.f;
    if (k < 1024) {
        if (k < wRows) val = W[(size_t)k * GATES + col];
    } else {
        int ku = k - 1024;
        if (ku < uRows) val = U[(size_t)ku * GATES + col];
    }
    dst[e] = (__bf16)val;
}

// x [B][T][D] f32  ->  xbf [T][B][D] bf16  (time-major for per-step A tiles)
__global__ void xconv_kernel(const float* __restrict__ x, __bf16* __restrict__ xbf) {
    size_t e = (size_t)blockIdx.x * 256 + threadIdx.x;
    if (e >= (size_t)T_STEPS * BATCH * IN_DIM) return;
    int d = (int)(e & 1023);
    int b = (int)((e >> 10) & 63);
    int t = (int)(e >> 16);
    xbf[e] = (__bf16)x[((size_t)b * T_STEPS + t) * IN_DIM + d];
}

__global__ void init_state_kernel(__bf16* h1a, __bf16* h1b, __bf16* h2a, __bf16* h2b,
                                  float* c1, float* c2) {
    int i = blockIdx.x * 256 + threadIdx.x;
    if (i >= BATCH * HPAD) return;
    __bf16 z = (__bf16)0.f;
    h1a[i] = z; h1b[i] = z; h2a[i] = z; h2b[i] = z;
    c1[i] = 0.f; c2[i] = 0.f;
}

// ---------------------------------------------------------------------------
// One LSTM layer step, fused GEMM + gates.
// Block = TWO N-tiles (32 gate-interleaved cols = 8 hidden units), 4 waves;
// wave w owns batch rows [16w, 16w+16) and keeps two f32 accumulators.
// K-loop split into x-half / h-half with pointer-increment addressing; the
// shared A fragment feeds two back-to-back identical WMMAs (reuse_a on the
// second), then the LSTM pointwise is applied in-block via an LDS z-tile.
// ---------------------------------------------------------------------------
__global__ __launch_bounds__(128) void lstm_layer_kernel(
    const __bf16* __restrict__ A0,    // first  K half [64 x 1024] (x_t or h1)
    const __bf16* __restrict__ A1,    // second K half [64 x 1024] (h_prev)
    const __bf16* __restrict__ Bpack, // packed weights
    const float*  __restrict__ bias,  // [4000] gate-major
    float*        __restrict__ cbuf,  // [64 x 1024] in-place cell state
    __bf16*       __restrict__ hout)  // [64 x 1024] new hidden
{
    __shared__ float zt[4][16][32];
    const int nt0  = blockIdx.x * 2;
    const int w    = threadIdx.x >> 5;
    const int lane = threadIdx.x & 31;
    const int rlo  = lane & 15;
    const int arow = w * 16 + rlo;           // A-matrix row (lanes 0-15 == 16-31)
    const int koff = (lane < 16) ? 0 : 8;

    v8f acc0 = {}, acc1 = {};
    const __bf16* b0Ptr = Bpack + (((size_t)nt0 * KT) << 9) + lane * 16;
    const __bf16* b1Ptr = b0Ptr + ((size_t)KT << 9);

    for (int half = 0; half < 2; ++half) {
        const __bf16* aPtr = (half ? A1 : A0) + (size_t)arow * 1024 + koff;
        #pragma unroll 4
        for (int kk = 0; kk < 32; ++kk) {
            v8bf a_lo = *(const v8bf*)aPtr;
            v8bf a_hi = *(const v8bf*)(aPtr + 16);
            v8bf b0_lo = *(const v8bf*)b0Ptr;
            v8bf b0_hi = *(const v8bf*)(b0Ptr + 8);
            v8bf b1_lo = *(const v8bf*)b1Ptr;
            v8bf b1_hi = *(const v8bf*)(b1Ptr + 8);
            __builtin_prefetch(b0Ptr + FRAG_ELEMS, 0, 3);   // keep weights near (L2-resident)
            __builtin_prefetch(b1Ptr + FRAG_ELEMS, 0, 3);
            v16bf a, b0, b1;
            #pragma unroll
            for (int i = 0; i < 8; ++i) {
                a[i]  = a_lo[i];  a[i + 8]  = a_hi[i];
                b0[i] = b0_lo[i]; b0[i + 8] = b0_hi[i];
                b1[i] = b1_lo[i]; b1[i + 8] = b1_hi[i];
            }
            acc0 = __builtin_amdgcn_wmma_f32_16x16x32_bf16(false, a, false, b0,
                                                           (short)0, acc0, false, false);
            // identical instruction as previous -> A-operand reuse hint is legal
            acc1 = __builtin_amdgcn_wmma_f32_16x16x32_bf16(false, a, false, b1,
                                                           (short)0, acc1, true, false);
            aPtr  += 32;
            b0Ptr += FRAG_ELEMS;
            b1Ptr += FRAG_ELEMS;
        }
    }

    // Stage both 16x16 z tiles: lane holds col rlo, rows {0..7}|{8..15}
    const int rbase = (lane < 16) ? 0 : 8;
    #pragma unroll
    for (int i = 0; i < 8; ++i) {
        zt[w][rbase + i][rlo]      = acc0[i];
        zt[w][rbase + i][16 + rlo] = acc1[i];
    }
    __syncthreads();

    // Pointwise LSTM: 16 rows x 8 hidden units per wave -> 4 per lane
    #pragma unroll
    for (int it = 0; it < 4; ++it) {
        int q = it * 32 + lane;   // 0..127
        int r = q >> 3;           // row in tile
        int u = q & 7;            // unit in block (0..7)
        int j = blockIdx.x * 8 + u;   // hidden index 0..999
        int grow = w * 16 + r;        // batch row
        float zi = zt[w][r][u * 4 + 0] + bias[j];
        float zf = zt[w][r][u * 4 + 1] + bias[HID + j];
        float zg = zt[w][r][u * 4 + 2] + bias[2 * HID + j];
        float zo = zt[w][r][u * 4 + 3] + bias[3 * HID + j];
        float ig = 1.f / (1.f + __expf(-zi));
        float fg = 1.f / (1.f + __expf(-zf));
        float og = 1.f / (1.f + __expf(-zo));
        float gg = tanhf(zg);
        size_t idx = (size_t)grow * HPAD + j;
        float cn = fg * cbuf[idx] + ig * gg;
        cbuf[idx] = cn;
        hout[idx] = (__bf16)(og * tanhf(cn));
    }
}

// out[b][o] = h2[b][:] . Wo[:][o] + bo[o]   (one wave per output scalar)
__global__ void out_proj_kernel(const __bf16* __restrict__ h2,
                                const float* __restrict__ Wo,
                                const float* __restrict__ bo,
                                float* __restrict__ out) {
    int b = blockIdx.x / 6, o = blockIdx.x % 6;
    int lane = threadIdx.x;
    float s = 0.f;
    for (int j = lane; j < HID; j += 32)
        s += (float)h2[(size_t)b * HPAD + j] * Wo[(size_t)j * 6 + o];
    #pragma unroll
    for (int off = 16; off > 0; off >>= 1)
        s += __shfl_xor(s, off, 32);
    if (lane == 0) out[b * 6 + o] = s + bo[o];
}

// ---------------------------------------------------------------------------
extern "C" void kernel_launch(void* const* d_in, const int* in_sizes, int n_in,
                              void* d_out, int out_size, void* d_ws, size_t ws_size,
                              hipStream_t stream) {
    (void)in_sizes; (void)n_in; (void)out_size; (void)ws_size;
    const float* x  = (const float*)d_in[0];
    const float* W1 = (const float*)d_in[1];
    const float* U1 = (const float*)d_in[2];
    const float* b1 = (const float*)d_in[3];
    const float* W2 = (const float*)d_in[4];
    const float* U2 = (const float*)d_in[5];
    const float* b2 = (const float*)d_in[6];
    const float* Wo = (const float*)d_in[7];
    const float* bo = (const float*)d_in[8];

    char* ws = (char*)d_ws;
    size_t off = 0;
    __bf16* wu1 = (__bf16*)(ws + off); off += PACK_ELEMS * 2;                 // 16,384,000
    __bf16* wu2 = (__bf16*)(ws + off); off += PACK_ELEMS * 2;                 // 16,384,000
    __bf16* xbf = (__bf16*)(ws + off); off += (size_t)T_STEPS * BATCH * IN_DIM * 2; // 16,777,216
    __bf16* h1a = (__bf16*)(ws + off); off += BATCH * HPAD * 2;
    __bf16* h1b = (__bf16*)(ws + off); off += BATCH * HPAD * 2;
    __bf16* h2a = (__bf16*)(ws + off); off += BATCH * HPAD * 2;
    __bf16* h2b = (__bf16*)(ws + off); off += BATCH * HPAD * 2;
    float*  c1  = (float*)(ws + off);  off += BATCH * HPAD * 4;
    float*  c2  = (float*)(ws + off);  off += BATCH * HPAD * 4;

    const int packBlocks = (int)(PACK_ELEMS / 256);                 // 32000
    pack_wu_kernel<<<packBlocks, 256, 0, stream>>>(W1, 1024, U1, HID, wu1);
    pack_wu_kernel<<<packBlocks, 256, 0, stream>>>(W2, HID,  U2, HID, wu2);
    xconv_kernel<<<(T_STEPS * BATCH * IN_DIM) / 256, 256, 0, stream>>>(x, xbf);
    init_state_kernel<<<(BATCH * HPAD + 255) / 256, 256, 0, stream>>>(h1a, h1b, h2a, h2b, c1, c2);

    for (int t = 0; t < T_STEPS; ++t) {
        __bf16* h1p = (t & 1) ? h1b : h1a;
        __bf16* h1n = (t & 1) ? h1a : h1b;
        __bf16* h2p = (t & 1) ? h2b : h2a;
        __bf16* h2n = (t & 1) ? h2a : h2b;
        const __bf16* xt = xbf + (size_t)t * BATCH * IN_DIM;
        lstm_layer_kernel<<<NT / 2, 128, 0, stream>>>(xt,  h1p, wu1, b1, c1, h1n);
        lstm_layer_kernel<<<NT / 2, 128, 0, stream>>>(h1n, h2p, wu2, b2, c2, h2n);
    }
    // after t = 127 (odd), the final h2 lives in h2a
    out_proj_kernel<<<BATCH * 6, 32, 0, stream>>>(h2a, Wo, bo, (float*)d_out);
}
